// Non_local_2542620639423
// MI455X (gfx1250) — matrix operand order repaired and checked by reading, not compiled, not run on hardware
//
#include <hip/hip_runtime.h>

// CDNA5 / gfx1250 non-local block. wave32, WMMA 16x16x32 f16 with hi/lo split
// for ~f32 accuracy on the channel contractions. Memory-bound workload:
// phase1 reads x once (151 MB), phase2 re-reads x (L2-resident: 151MB < 192MB L2)
// and writes z. ~302 MB of HBM traffic ~= 13 us at 23.3 TB/s.

typedef __attribute__((ext_vector_type(16))) _Float16 v16h;
typedef __attribute__((ext_vector_type(8)))  float    v8f;

#define B_   32
#define C_   256
#define N_   4608
#define NT4  (N_ / 4)          // 1152
#define TPW  4                 // n-tiles (of 16) per wave
#define BLKX 9                 // blocks along n: 9 * (8 waves * 4 tiles * 16 n) = 4608
#define PPB  (BLKX * 8)        // partial sums per batch = 72

// ---------------------------------------------------------------------------
// Phase 1: per wave, compute D(16n x 16col) = A(16n x 32c) * Bw(32c x 16col)
// accumulated over 8 chunks of C=256. Columns: 0=g, 1=theta, 2=phi, 3..15=0.
__global__ __launch_bounds__(256) void nl_phase1(
    const float* __restrict__ x,
    const float* __restrict__ g_w, const float* __restrict__ g_b,
    const float* __restrict__ theta_w, const float* __restrict__ theta_b,
    const float* __restrict__ phi_w, const float* __restrict__ phi_b,
    float* __restrict__ theta_out, float* __restrict__ partials)
{
    const int tid   = threadIdx.x;
    const int wave  = tid >> 5;
    const int lane  = tid & 31;
    const int lm    = lane & 15;  // A: n-row within tile; B/D: output column
    const int half  = lane >> 4;
    const int b     = blockIdx.y;
    const int nbase = blockIdx.x * (8 * TPW * 16) + wave * 16;

    v8f acc[TPW];
    #pragma unroll
    for (int t = 0; t < TPW; ++t) acc[t] = (v8f){0,0,0,0,0,0,0,0};

    const size_t xb = (size_t)b * C_ * N_;

    for (int ck = 0; ck < 8; ++ck) {
        const int c0 = ck * 32;

        // B fragment (32x16 f16): lanes 0-15 hold K=0..7,16..23 (2/VGPR), lanes 16-31 K=8..15,24..31.
        v16h bhi, blo;
        #pragma unroll
        for (int e = 0; e < 16; ++e) {
            const int K = ((e >> 3) << 4) + (e & 7) + half * 8;
            float wv = 0.0f;
            if (lm == 0)      wv = g_w[c0 + K];
            else if (lm == 1) wv = theta_w[c0 + K];
            else if (lm == 2) wv = phi_w[c0 + K];
            const _Float16 wh = (_Float16)wv;
            bhi[e] = wh;
            blo[e] = (_Float16)(wv - (float)wh);
        }

        #pragma unroll
        for (int t = 0; t < TPW; ++t) {
            const int n0 = nbase + t * 128;
            // A fragment (16x32 f16): lane half selects K bank per ISA 7.12.2.
            v16h ahi, alo;
            #pragma unroll
            for (int e = 0; e < 16; ++e) {
                const int K = ((e >> 3) << 4) + (e & 7) + half * 8;
                const float xv = x[xb + (size_t)(c0 + K) * N_ + (n0 + lm)];
                const _Float16 xh = (_Float16)xv;
                ahi[e] = xh;
                alo[e] = (_Float16)(xv - (float)xh);
            }
            // hi*hi + lo*hi + hi*lo: near-f32 precision, f32 accumulate.
            acc[t] = __builtin_amdgcn_wmma_f32_16x16x32_f16(false, ahi, false, bhi, (short)0, acc[t], false, false);
            acc[t] = __builtin_amdgcn_wmma_f32_16x16x32_f16(false, alo, false, bhi, (short)0, acc[t], false, false);
            acc[t] = __builtin_amdgcn_wmma_f32_16x16x32_f16(false, ahi, false, blo, (short)0, acc[t], false, false);
        }
    }

    // D layout: reg r, lane l -> row n = r + 8*(l>>4), col = l&15.
    const float gb = g_b[0], tb = theta_b[0], pb = phi_b[0];
    float ps = 0.0f;
    #pragma unroll
    for (int t = 0; t < TPW; ++t) {
        const int n0 = nbase + t * 128;
        if (lm == 1) {  // theta column -> stash for phase 2
            #pragma unroll
            for (int r = 0; r < 8; ++r)
                theta_out[(size_t)b * N_ + n0 + 8 * half + r] = acc[t][r] + tb;
        }
        // phi*g partial: cols 0 and 2 live in lanes {0,2} and {16,18}.
        #pragma unroll
        for (int h = 0; h < 2; ++h) {
            #pragma unroll
            for (int r = 0; r < 8; ++r) {
                const float gv = __shfl(acc[t][r], h * 16 + 0, 32) + gb;
                const float pv = __shfl(acc[t][r], h * 16 + 2, 32) + pb;
                ps += gv * pv;  // uniform across lanes
            }
        }
    }
    if (lane == 0)
        partials[b * PPB + blockIdx.x * 8 + wave] = ps;
}

// ---------------------------------------------------------------------------
// Deterministic per-batch reduction: s[b] = sum(partials)/N
__global__ __launch_bounds__(128) void nl_reduce(
    const float* __restrict__ partials, float* __restrict__ s_final)
{
    __shared__ float sm[128];
    const int b = blockIdx.x, t = threadIdx.x;
    sm[t] = (t < PPB) ? partials[b * PPB + t] : 0.0f;
    __syncthreads();
    for (int off = 64; off > 0; off >>= 1) {
        if (t < off) sm[t] += sm[t + off];
        __syncthreads();
    }
    if (t == 0) s_final[b] = sm[0] * (1.0f / (float)N_);
}

// ---------------------------------------------------------------------------
// Phase 2: z = (theta*s*W_w + W_b - mu)*gamma/sqrt(var+eps) + beta + x
__global__ __launch_bounds__(256) void nl_phase2(
    const float* __restrict__ x,
    const float* __restrict__ W_w, const float* __restrict__ W_b,
    const float* __restrict__ bn_gamma, const float* __restrict__ bn_beta,
    const float* __restrict__ bn_mean, const float* __restrict__ bn_var,
    const float* __restrict__ theta, const float* __restrict__ s_final,
    float* __restrict__ out)
{
    const int i4 = blockIdx.x * 256 + threadIdx.x;   // < B*C*N/4
    const int n4 = i4 % NT4;
    const int bc = i4 / NT4;
    const int c  = bc % C_;
    const int b  = bc / C_;

    const float s    = s_final[b];
    const float istd = bn_gamma[c] * __frsqrt_rn(bn_var[c] + 1e-5f);
    const float k    = s * W_w[c] * istd;
    const float base = (W_b[c] - bn_mean[c]) * istd + bn_beta[c];

    const float4 th = reinterpret_cast<const float4*>(theta)[b * NT4 + n4];
    const float4 xv = reinterpret_cast<const float4*>(x)[i4];
    float4 z;
    z.x = fmaf(th.x, k, base) + xv.x;
    z.y = fmaf(th.y, k, base) + xv.y;
    z.z = fmaf(th.z, k, base) + xv.z;
    z.w = fmaf(th.w, k, base) + xv.w;
    reinterpret_cast<float4*>(out)[i4] = z;
}

// ---------------------------------------------------------------------------
extern "C" void kernel_launch(void* const* d_in, const int* in_sizes, int n_in,
                              void* d_out, int out_size, void* d_ws, size_t ws_size,
                              hipStream_t stream)
{
    (void)in_sizes; (void)n_in; (void)out_size; (void)ws_size;
    const float* x       = (const float*)d_in[0];
    const float* g_w     = (const float*)d_in[1];
    const float* g_b     = (const float*)d_in[2];
    const float* theta_w = (const float*)d_in[3];
    const float* theta_b = (const float*)d_in[4];
    const float* phi_w   = (const float*)d_in[5];
    const float* phi_b   = (const float*)d_in[6];
    const float* W_w     = (const float*)d_in[7];
    const float* W_b     = (const float*)d_in[8];
    const float* bn_g    = (const float*)d_in[9];
    const float* bn_b    = (const float*)d_in[10];
    const float* bn_m    = (const float*)d_in[11];
    const float* bn_v    = (const float*)d_in[12];

    float* ws       = (float*)d_ws;
    float* theta    = ws;                          // B*N floats
    float* partials = ws + (size_t)B_ * N_;        // B*PPB floats
    float* s_final  = partials + B_ * PPB;         // B floats

    nl_phase1<<<dim3(BLKX, B_), 256, 0, stream>>>(
        x, g_w, g_b, theta_w, theta_b, phi_w, phi_b, theta, partials);
    nl_reduce<<<B_, 128, 0, stream>>>(partials, s_final);
    nl_phase2<<<(B_ * C_ * N_ / 4) / 256, 256, 0, stream>>>(
        x, W_w, W_b, bn_g, bn_b, bn_m, bn_v, theta, s_final, (float*)d_out);
}
